// SSMBlock_28381143892541
// MI455X (gfx1250) — matrix operand order repaired and checked by reading, not compiled, not run on hardware
//
#include <hip/hip_runtime.h>

// ---------------- problem constants ----------------
#define BB    4
#define SS    4096
#define DD    512
#define NNs   512
#define MM    (BB * SS)          // 16384
#define CHUNK 256                // scan chunk length
#define NCH   (SS / CHUNK)       // 16 chunks

typedef __bf16 bf16_t;
typedef bf16_t v16bf __attribute__((ext_vector_type(16)));
typedef float  v8f   __attribute__((ext_vector_type(8)));
typedef int    v4i_t __attribute__((vector_size(16)));   // int4, matches builtin proto

#if __has_builtin(__builtin_amdgcn_global_load_async_to_lds_b128)
#define SSM_HAS_ASYNC 1
#else
#define SSM_HAS_ASYNC 0
#endif

// 16-byte global -> LDS copy. Async (ASYNCcnt-tracked, no VGPR staging) when the
// gfx1250 builtin is available; otherwise classic register staging.
__device__ __forceinline__ void ssm_copy16(const float* g, float* l) {
#if SSM_HAS_ASYNC
    __builtin_amdgcn_global_load_async_to_lds_b128(
        (__attribute__((address_space(1))) v4i_t*)g,
        (__attribute__((address_space(3))) v4i_t*)l, 0, 0);
#else
    *(float4*)l = *(const float4*)g;
#endif
}

__device__ __forceinline__ void ssm_wait_async() {
#if SSM_HAS_ASYNC
#if __has_builtin(__builtin_amdgcn_s_wait_asynccnt)
    __builtin_amdgcn_s_wait_asynccnt(0);
#else
    asm volatile("s_wait_asynccnt 0x0" ::: "memory");
#endif
#endif
}

// pack two float4 runs into 8 consecutive bf16 fragment elements
__device__ __forceinline__ void pack8(v16bf& v, int base, float4 a, float4 b) {
    v[base + 0] = (bf16_t)a.x; v[base + 1] = (bf16_t)a.y;
    v[base + 2] = (bf16_t)a.z; v[base + 3] = (bf16_t)a.w;
    v[base + 4] = (bf16_t)b.x; v[base + 5] = (bf16_t)b.y;
    v[base + 6] = (bf16_t)b.z; v[base + 7] = (bf16_t)b.w;
}

// ---------------------------------------------------------------------------
// Tiled bf16-WMMA GEMM:  out[M,Nout] = A[M,K] * B  (+ epilogue)
//   TRANSB=true : B source row-major [Nout,K]  (computes A @ Bsrc^T)
//   TRANSB=false: B source row-major [K,Nout]  (computes A @ Bsrc)
//   EPI 0: out = v + bias[n]
//   EPI 1: out = silu(v + bias[n])
//   EPI 2: out = v + dparam[n] * aux[m,n]
// 256 threads (8 waves). Block tile 128(M) x 64(N), BK=32, double-buffered LDS
// filled by GLOBAL_LOAD_ASYNC_TO_LDS_B128. Wave (wm,wn) owns 32x32 = 2x2 WMMAs.
// ---------------------------------------------------------------------------
template <int EPI, bool TRANSB>
__global__ __launch_bounds__(256)
void ssm_gemm(const float* __restrict__ A, const float* __restrict__ Bsrc,
              const float* __restrict__ bias, const float* __restrict__ dparam,
              const float* __restrict__ aux, float* __restrict__ out,
              int M, int Nout, int K)
{
    __shared__ __align__(16) float sA[2][128 * 32];   // [m][k]
    __shared__ __align__(16) float sB[2][2048];       // NN: [k 32][n 64]; TRANSB: [n 64][k 32]

    const int t    = threadIdx.x;
    const int lane = t & 31;
    const int wave = t >> 5;
    const int wm   = wave >> 1;            // 0..3
    const int wn   = wave & 1;             // 0..1
    const int m0   = blockIdx.y * 128;
    const int n0   = blockIdx.x * 64;

    v8f acc[2][2];
    #pragma unroll
    for (int i = 0; i < 2; ++i)
        #pragma unroll
        for (int j = 0; j < 2; ++j)
            #pragma unroll
            for (int e = 0; e < 8; ++e) acc[i][j][e] = 0.f;

    const int mrow = lane & 15;
    const int kb   = (lane & 16) ? 8 : 0;  // A-frag K base per ISA layout

    // issue async copies for one 128x32 A tile + 32x64 B tile into buffer `buf`
    auto stage = [&](int buf, int kk) {
        #pragma unroll
        for (int i = 0; i < 4; ++i) {              // 1024 float4 slots, 4 per thread
            int fi = i * 256 + t;
            int r = fi >> 3, c4 = fi & 7;
            ssm_copy16(A + (size_t)(m0 + r) * K + kk + c4 * 4,
                       &sA[buf][r * 32 + c4 * 4]);
        }
        if (TRANSB) {                              // land verbatim as [n][k]
            #pragma unroll
            for (int i = 0; i < 2; ++i) {
                int fi = i * 256 + t;
                int r = fi >> 3, c4 = fi & 7;      // r = n (0..63), c4 over k
                ssm_copy16(Bsrc + (size_t)(n0 + r) * K + kk + c4 * 4,
                           &sB[buf][r * 32 + c4 * 4]);
            }
        } else {                                   // land verbatim as [k][n]
            #pragma unroll
            for (int i = 0; i < 2; ++i) {
                int fi = i * 256 + t;
                int r = fi >> 4, c4 = fi & 15;     // r = k (0..31), c4 over n
                ssm_copy16(Bsrc + (size_t)(kk + r) * Nout + n0 + c4 * 4,
                           &sB[buf][r * 64 + c4 * 4]);
            }
        }
    };

    stage(0, 0);
    ssm_wait_async();
    __syncthreads();

    for (int kk = 0; kk < K; kk += 32) {
        const int cur = (kk >> 5) & 1;
        if (kk + 32 < K) stage(cur ^ 1, kk + 32);   // async fill of next buffer

        // ---- build fragments (ISA 7.12.2 layouts), fp32 -> bf16 ----
        v16bf afr[2], bfr[2];
        #pragma unroll
        for (int i = 0; i < 2; ++i) {
            const float* p = &sA[cur][(wm * 32 + i * 16 + mrow) * 32];
            float4 x0 = *(const float4*)(p + kb);
            float4 x1 = *(const float4*)(p + kb + 4);
            float4 x2 = *(const float4*)(p + kb + 16);
            float4 x3 = *(const float4*)(p + kb + 20);
            v16bf a;
            pack8(a, 0, x0, x1);
            pack8(a, 8, x2, x3);
            afr[i] = a;
        }
        if (TRANSB) {
            // sB is [n][k]; frag element (n=lane%16, k=table order like A)
            #pragma unroll
            for (int j = 0; j < 2; ++j) {
                const float* p = &sB[cur][(wn * 32 + j * 16 + mrow) * 32];
                v16bf b;
                #pragma unroll
                for (int e = 0; e < 16; ++e) b[e] = (bf16_t)p[kb + (e & 7) + ((e >> 3) << 4)];
                bfr[j] = b;
            }
        } else {
            // sB is [k][n]; frag element (n=lane%16, k=table order like A)
            #pragma unroll
            for (int j = 0; j < 2; ++j) {
                const float* p = &sB[cur][wn * 32 + j * 16 + mrow];
                v16bf b;
                #pragma unroll
                for (int e = 0; e < 16; ++e) b[e] = (bf16_t)p[(kb + (e & 7) + ((e >> 3) << 4)) * 64];
                bfr[j] = b;
            }
        }

        #pragma unroll
        for (int i = 0; i < 2; ++i)
            #pragma unroll
            for (int j = 0; j < 2; ++j)
                acc[i][j] = __builtin_amdgcn_wmma_f32_16x16x32_bf16(
                    false, afr[i], false, bfr[j], (short)0, acc[i][j], false, false);

        ssm_wait_async();      // next buffer landed (this wave)
        __syncthreads();       // ... and for all waves; cur readers also done
    }

    // ---- epilogue + store (C layout: vgpr r -> M = r + 8*(lane>=16), N = lane&15) ----
    const int mofs = (lane >> 4) << 3;
    const int ncol = lane & 15;
    #pragma unroll
    for (int i = 0; i < 2; ++i) {
        #pragma unroll
        for (int j = 0; j < 2; ++j) {
            int mb = m0 + wm * 32 + i * 16 + mofs;
            int nb = n0 + wn * 32 + j * 16 + ncol;
            #pragma unroll
            for (int r = 0; r < 8; ++r) {
                int m = mb + r;
                float v = acc[i][j][r];
                if (EPI == 1) { v += bias[nb]; v = v / (1.f + __expf(-v)); }
                else if (EPI == 0) { v += bias[nb]; }
                else { v += dparam[nb] * aux[(size_t)m * Nout + nb]; }
                out[(size_t)m * Nout + nb] = v;
            }
        }
    }
}

// ---------------- decay[n] = exp(-mean_k exp(A_log[n,k])) ----------------
__global__ void ssm_decay(const float* __restrict__ Alog, float* __restrict__ decay)
{
    int j = blockIdx.x * blockDim.x + threadIdx.x;
    if (j >= DD) return;
    float acc = 0.f;
    for (int k = 0; k < NNs; ++k) acc += __expf(Alog[(size_t)j * NNs + k]);
    decay[j] = __expf(-acc / (float)NNs);
}

// ---- scan pass 1: per-chunk local scan (zero init), in place over s ----
__global__ void ssm_scan_chunk(float* __restrict__ s, const float* __restrict__ decay,
                               float* __restrict__ partial)
{
    int gid = blockIdx.x * blockDim.x + threadIdx.x;   // BB*NCH*NNs threads
    int n = gid % NNs;
    int c = (gid / NNs) % NCH;
    int b = gid / (NNs * NCH);
    float d = decay[n];
    float st = 0.f;
    size_t base = ((size_t)b * SS + (size_t)c * CHUNK) * NNs + n;
    for (int t = 0; t < CHUNK; ++t) {
        st = d * st + s[base + (size_t)t * NNs];
        s[base + (size_t)t * NNs] = st;
    }
    partial[((size_t)b * NCH + c) * NNs + n] = st;
}

// ---- scan pass 2: prefix over chunk partials (carry in = state0) ----
__global__ void ssm_scan_carry(const float* __restrict__ partial,
                               const float* __restrict__ decay,
                               const float* __restrict__ state0,
                               float* __restrict__ carry,
                               float* __restrict__ statef)
{
    int gid = blockIdx.x * blockDim.x + threadIdx.x;   // BB*NNs threads
    int n = gid % NNs;
    int b = gid / NNs;
    float d = decay[n];
    float dp = 1.f;
    for (int i = 0; i < CHUNK; ++i) dp *= d;           // d^CHUNK
    float cv = state0[(size_t)b * NNs + n];
    for (int c = 0; c < NCH; ++c) {
        carry[((size_t)b * NCH + c) * NNs + n] = cv;   // state entering chunk c
        cv = dp * cv + partial[((size_t)b * NCH + c) * NNs + n];
    }
    statef[(size_t)b * NNs + n] = cv;                  // final state -> d_out tail
}

// ---- scan pass 3: states[t] += d^(t_local+1) * carry(chunk) ----
__global__ void ssm_scan_fix(float* __restrict__ s, const float* __restrict__ decay,
                             const float* __restrict__ carry)
{
    int gid = blockIdx.x * blockDim.x + threadIdx.x;   // BB*NCH*NNs threads
    int n = gid % NNs;
    int c = (gid / NNs) % NCH;
    int b = gid / (NNs * NCH);
    float d = decay[n];
    float cv = carry[((size_t)b * NCH + c) * NNs + n];
    float p = d;
    size_t base = ((size_t)b * SS + (size_t)c * CHUNK) * NNs + n;
    for (int t = 0; t < CHUNK; ++t) {
        s[base + (size_t)t * NNs] += p * cv;
        p *= d;
    }
}

// ---------------------------------------------------------------------------
extern "C" void kernel_launch(void* const* d_in, const int* in_sizes, int n_in,
                              void* d_out, int out_size, void* d_ws, size_t ws_size,
                              hipStream_t stream)
{
    const float* x      = (const float*)d_in[0];
    const float* state0 = (const float*)d_in[1];
    const float* Wg     = (const float*)d_in[2];
    const float* bg     = (const float*)d_in[3];
    const float* Wsp    = (const float*)d_in[4];
    const float* bsp    = (const float*)d_in[5];
    const float* Wo     = (const float*)d_in[6];
    const float* bo     = (const float*)d_in[7];
    const float* Alog   = (const float*)d_in[8];
    const float* Dp     = (const float*)d_in[9];
    float* out = (float*)d_out;

    float* xgate = (float*)d_ws;                        // MM*DD
    float* sbuf  = xgate + (size_t)MM * DD;             // MM*NNs (becomes states in place)
    float* ybuf  = sbuf  + (size_t)MM * NNs;            // MM*DD
    float* decay = ybuf  + (size_t)MM * DD;             // DD
    float* part  = decay + DD;                          // BB*NCH*NNs
    float* carry = part  + (size_t)BB * NCH * NNs;      // BB*NCH*NNs

    dim3 blk(256);
    dim3 gg(DD / 64, MM / 128);                         // (8, 128) blocks

    // 1) x_gate = silu(x @ Wg^T + bg)
    ssm_gemm<1, true><<<gg, blk, 0, stream>>>(x, Wg, bg, nullptr, nullptr, xgate, MM, DD, DD);
    // 2) s = x_gate @ Wsp^T + bsp
    ssm_gemm<0, true><<<gg, blk, 0, stream>>>(xgate, Wsp, bsp, nullptr, nullptr, sbuf, MM, NNs, DD);

    // 3) chunked diagonal scan (in place: sbuf -> states), final state to d_out tail
    ssm_decay<<<dim3((DD + 255) / 256), blk, 0, stream>>>(Alog, decay);
    ssm_scan_chunk<<<dim3((BB * NCH * NNs) / 256), blk, 0, stream>>>(sbuf, decay, part);
    float* statef = out + (size_t)MM * DD;
    ssm_scan_carry<<<dim3((BB * NNs) / 256), blk, 0, stream>>>(part, decay, state0, carry, statef);
    ssm_scan_fix<<<dim3((BB * NCH * NNs) / 256), blk, 0, stream>>>(sbuf, decay, carry);

    // 4) y = states @ Wsp + Dp * x_gate
    ssm_gemm<2, false><<<gg, blk, 0, stream>>>(sbuf, Wsp, nullptr, Dp, xgate, ybuf, MM, DD, NNs);
    // 5) out = y @ Wo^T + bo
    ssm_gemm<0, true><<<gg, blk, 0, stream>>>(ybuf, Wo, bo, nullptr, nullptr, out, MM, DD, DD);
}
